// MultiKPGenerator_63831803953433
// MI455X (gfx1250) — matrix operand order repaired and checked by reading, not compiled, not run on hardware
//
#include <hip/hip_runtime.h>
#include <cstdint>
#include <cstddef>

// ---------------- problem constants ----------------
constexpr int B = 32, C = 1024, N = 2048, K = 64;
constexpr int CIN = 2 * C + 128;          // 2176
constexpr int NP_DW = C / 2;              // 512 dwords per bf16 point row

// ---------------- WMMA / vector types ----------------
typedef __attribute__((ext_vector_type(16))) __bf16   v16bf;
typedef __attribute__((ext_vector_type(8)))  float    v8f;
typedef __attribute__((ext_vector_type(4)))  unsigned v4u;

union FragAB { v16bf bf; unsigned u[8]; v4u q4[2]; };

// ---------------- scalar helpers ----------------
static __device__ __forceinline__ unsigned short f2bf(float f) {
  union { float f; unsigned u; } v; v.f = f;
  unsigned r = 0x7FFFu + ((v.u >> 16) & 1u);   // round-to-nearest-even
  v.u += r;
  return (unsigned short)(v.u >> 16);
}
static __device__ __forceinline__ unsigned pk2(float lo, float hi) {
  return (unsigned)f2bf(lo) | ((unsigned)f2bf(hi) << 16);
}
static __device__ __forceinline__ float bflo(unsigned u) {
  union { unsigned u; float f; } v; v.u = u << 16; return v.f;
}
static __device__ __forceinline__ float bfhi(unsigned u) {
  union { unsigned u; float f; } v; v.u = u & 0xFFFF0000u; return v.f;
}

// =====================================================================
// K0: (B,C,N) f32 -> (B,N,C) bf16, LDS-tiled transpose
// =====================================================================
__global__ void k_transpose(const float* __restrict__ pf, unsigned short* __restrict__ xb) {
  __shared__ float t[32][33];
  int b = blockIdx.z, n0 = blockIdx.x * 32, c0 = blockIdx.y * 32;
  const float* src = pf + (size_t)b * C * N;
  for (int i = 0; i < 32; i += 8)
    t[threadIdx.y + i][threadIdx.x] = src[(size_t)(c0 + threadIdx.y + i) * N + n0 + threadIdx.x];
  __syncthreads();
  unsigned short* dst = xb + (size_t)b * N * C;
  for (int i = 0; i < 32; i += 8)
    dst[(size_t)(n0 + threadIdx.y + i) * C + c0 + threadIdx.x] = f2bf(t[threadIdx.x][threadIdx.y + i]);
}

// K1: flat f32 -> bf16
__global__ void k_cvt_bf16(const float* __restrict__ in, unsigned short* __restrict__ out, int n) {
  int i = blockIdx.x * blockDim.x + threadIdx.x;
  if (i < n) out[i] = f2bf(in[i]);
}

// =====================================================================
// K2: pack ps1_w (1024 x 8192) f32 into fragment-blocked bf16-pair layout:
// block(nt,kc) of 256 dwords, addr = ((nt*32+kc)*256 + lane*8 + j);
// dword(lane,j) = rows {2*pr, 2*pr+1} packed, pr = kc*16 + j + 8*hi, col = nt*16+lm
// =====================================================================
__global__ void k_pack_wps(const float* __restrict__ in, unsigned* __restrict__ out) {
  int d = blockIdx.x * blockDim.x + threadIdx.x;
  if (d >= (C / 2) * 8192) return;
  int j = d & 7, lane = (d >> 3) & 31, blk = d >> 8;
  int kc = blk & 31, nt = blk >> 5;
  int hi = lane >> 4, lm = lane & 15;
  int pr = kc * 16 + j + 8 * hi;
  int col = nt * 16 + lm;
  out[d] = pk2(in[(size_t)(2 * pr) * 8192 + col], in[(size_t)(2 * pr + 1) * 8192 + col]);
}

// K3: init FPS state
__global__ void k_init(float* __restrict__ dists, int* __restrict__ curidx, int* __restrict__ idxbuf) {
  int i = blockIdx.x * blockDim.x + threadIdx.x;
  if (i < B * N) dists[i] = 1e10f;
  if (i < B) { curidx[i] = 0; idxbuf[i * K] = 0; }   // scan emits farthest BEFORE update -> idx[:,0]=0
}

// =====================================================================
// K4: one FPS step. Channel sum is order-invariant, so each lane owns 16
// CONTIGUOUS dwords of the row: 4x global_load_b128 + 4x ds_load_b128.
// grid (16, B), block 256 (8 waves x 16 points)
// =====================================================================
__global__ void k_fps_dist(const unsigned* __restrict__ xb32, const int* __restrict__ curidx,
                           float* __restrict__ dists, float* __restrict__ partD, int* __restrict__ partI) {
  __shared__ unsigned cent[NP_DW];
  __shared__ float wbD[8];
  __shared__ int   wbI[8];
  int b = blockIdx.y, chunk = blockIdx.x;
  int ci = curidx[b];
  const unsigned* crow = xb32 + ((size_t)b * N + ci) * NP_DW;
  cent[threadIdx.x]       = crow[threadIdx.x];
  cent[threadIdx.x + 256] = crow[threadIdx.x + 256];
  __syncthreads();
  int w = threadIdx.x >> 5, lane = threadIdx.x & 31;
  const v4u* cv4 = (const v4u*)(cent + lane * 16);
  float bestD = -1.0f; int bestI = 0;
  for (int j = 0; j < 16; ++j) {
    int p = chunk * 128 + w * 16 + j;
    const unsigned* prow = xb32 + ((size_t)b * N + p) * NP_DW;
    if (j + 1 < 16) __builtin_prefetch(prow + NP_DW, 0, 1);  // -> global_prefetch_b8
    const v4u* pv = (const v4u*)(prow + lane * 16);
    float acc = 0.0f;
#pragma unroll
    for (int q = 0; q < 4; ++q) {
      v4u v = pv[q];
      v4u cc = cv4[q];
#pragma unroll
      for (int e = 0; e < 4; ++e) {
        float d0 = bflo(v[e]) - bflo(cc[e]);
        float d1 = bfhi(v[e]) - bfhi(cc[e]);
        acc = fmaf(d0, d0, acc);
        acc = fmaf(d1, d1, acc);
      }
    }
#pragma unroll
    for (int m = 16; m >= 1; m >>= 1) acc += __shfl_xor(acc, m, 32);
    if (lane == 0) {
      float d = fminf(dists[b * N + p], acc);
      dists[b * N + p] = d;
      if (d > bestD) { bestD = d; bestI = p; }       // strict > keeps first (argmax tie rule)
    }
  }
  if (lane == 0) { wbD[w] = bestD; wbI[w] = bestI; }
  __syncthreads();
  if (threadIdx.x == 0) {
    float bd = wbD[0]; int bi = wbI[0];
    for (int q = 1; q < 8; ++q) if (wbD[q] > bd) { bd = wbD[q]; bi = wbI[q]; }
    partD[b * 16 + chunk] = bd; partI[b * 16 + chunk] = bi;
  }
}

// K5: reduce 16 partials per batch -> idx[b][t], curidx[b]
__global__ void k_fps_argmax(const float* __restrict__ partD, const int* __restrict__ partI,
                             int* __restrict__ curidx, int* __restrict__ idxbuf, int t) {
  int b = blockIdx.x;
  if (threadIdx.x == 0) {
    float bd = partD[b * 16]; int bi = partI[b * 16];
    for (int q = 1; q < 16; ++q) { float d = partD[b * 16 + q]; if (d > bd) { bd = d; bi = partI[b * 16 + q]; } }
    idxbuf[b * K + t] = bi;
    curidx[b] = bi;
  }
}

// =====================================================================
// K6: x1 GEMM (32 x 1024)@(1024 x 8192) -> x1f (B,128,64) f32 (+bias).
// A & B fragments each = 2x global_load_b128.
// grid 32, block 256: 256 waves x 4 output tiles, 32 k-chunks of 32.
// =====================================================================
__global__ void k_gemm1(const unsigned short* __restrict__ gfb, const unsigned* __restrict__ wpsf,
                        const float* __restrict__ ps1_b, float* __restrict__ x1f) {
  int w = threadIdx.x >> 5, lane = threadIdx.x & 31;
  int gw = blockIdx.x * 8 + w;            // 0..255
  int mtile = gw & 1;
  int ntb = (gw >> 1) * 4;                // 4 ntiles per wave
  int hi = lane >> 4, lm = lane & 15;
  v8f acc[4] = {};
  const unsigned* arow = (const unsigned*)gfb + (size_t)(mtile * 16 + lm) * (C / 2);
  for (int kc = 0; kc < C / 32; ++kc) {
    FragAB a;
    a.q4[0] = *(const v4u*)(arow + kc * 16 + hi * 8);       // j=0..3
    a.q4[1] = *(const v4u*)(arow + kc * 16 + hi * 8 + 4);   // j=4..7
#pragma unroll
    for (int i = 0; i < 4; ++i) {
      const unsigned* bb = wpsf + ((size_t)(ntb + i) * 32 + kc) * 256 + lane * 8;
      FragAB bm;
      bm.q4[0] = *(const v4u*)(bb);
      bm.q4[1] = *(const v4u*)(bb + 4);
      acc[i] = __builtin_amdgcn_wmma_f32_16x16x32_bf16(false, a.bf, false, bm.bf,
                                                       (short)0, acc[i], false, false);
    }
  }
#pragma unroll
  for (int i = 0; i < 4; ++i) {
    int ncol = (ntb + i) * 16 + lm;
    int o = ncol >> 6, kk = ncol & 63;
    float bias = ps1_b[o];
#pragma unroll
    for (int r = 0; r < 8; ++r) {
      int bbat = mtile * 16 + r + 8 * hi;  // D layout: row = r + 8*hi, col = lane&15
      x1f[(size_t)bbat * 8192 + o * 64 + kk] = acc[i][r] + bias;
    }
  }
}

// =====================================================================
// K7: build catp directly in fragment-blocked layout:
// per batch, block(kc,nt) of 256 dwords at ((kc*4+nt)*256 + lane*8 + j);
// dword = cat rows {2pr, 2pr+1} bf16-packed at col, pr = kc*16+j+8*hi, col = nt*16+lm.
// cat rows: [0,128) = x1, [128,1152) = gathered point feat, [1152,2176) = gf.
// =====================================================================
__global__ void k_pack_cat(const float* __restrict__ x1f, const unsigned short* __restrict__ xb,
                           const unsigned short* __restrict__ gfb, const int* __restrict__ idxbuf,
                           unsigned* __restrict__ catp) {
  int i = blockIdx.x * blockDim.x + threadIdx.x;
  if (i >= B * (CIN / 2) * K) return;
  int d = i % ((CIN / 2) * K);
  int b = i / ((CIN / 2) * K);
  int j = d & 7, lane = (d >> 3) & 31, blk = d >> 8;
  int nt = blk & 3, kc = blk >> 2;
  int hi = lane >> 4, lm = lane & 15;
  int pr = kc * 16 + j + 8 * hi;
  int kk = nt * 16 + lm;
  int row = 2 * pr;
  unsigned val;
  if (row < 128) {
    val = pk2(x1f[(size_t)b * 8192 + row * 64 + kk], x1f[(size_t)b * 8192 + (row + 1) * 64 + kk]);
  } else if (row < 128 + C) {
    int c = row - 128;                                    // even
    int p = idxbuf[b * K + kk];
    val = ((const unsigned*)xb)[((size_t)b * N + p) * NP_DW + (c >> 1)];  // already a bf16 pair
  } else {
    int c = row - (128 + C);                              // even
    val = ((const unsigned*)gfb)[(size_t)b * (C / 2) + (c >> 1)];
  }
  catp[i] = val;
}

// =====================================================================
// K8: per-batch GEMM (128 x 2176)@(2176 x 64). A/B fragments = 2x b128 each.
// mode 0: m1c1 -> bias+ReLU, packed bf16 h1p; mode 1: m1sc -> f32 scf.
// grid B, block 256 (wave = mtile).
// =====================================================================
__global__ void k_gemm2(const unsigned short* __restrict__ wbf, const float* __restrict__ bias,
                        const unsigned* __restrict__ catp, unsigned* __restrict__ h1p,
                        float* __restrict__ scf, int mode) {
  int b = blockIdx.x;
  int w = threadIdx.x >> 5, lane = threadIdx.x & 31, hi = lane >> 4, lm = lane & 15;
  int mtile = w;
  v8f acc[4] = {};
  const unsigned* arow = (const unsigned*)wbf + (size_t)(mtile * 16 + lm) * (CIN / 2);
  const unsigned* cp = catp + (size_t)b * (CIN / 2) * K;
  for (int kc = 0; kc < CIN / 32; ++kc) {                 // 68 chunks
    FragAB a;
    a.q4[0] = *(const v4u*)(arow + kc * 16 + hi * 8);
    a.q4[1] = *(const v4u*)(arow + kc * 16 + hi * 8 + 4);
#pragma unroll
    for (int i = 0; i < 4; ++i) {
      const unsigned* bb = cp + (size_t)(kc * 4 + i) * 256 + lane * 8;
      FragAB bm;
      bm.q4[0] = *(const v4u*)(bb);
      bm.q4[1] = *(const v4u*)(bb + 4);
      acc[i] = __builtin_amdgcn_wmma_f32_16x16x32_bf16(false, a.bf, false, bm.bf,
                                                       (short)0, acc[i], false, false);
    }
  }
  if (mode == 0) {
#pragma unroll
    for (int i = 0; i < 4; ++i) {
      int col = i * 16 + lm;
#pragma unroll
      for (int q = 0; q < 4; ++q) {
        int row0 = mtile * 16 + 8 * hi + 2 * q;
        float v0 = fmaxf(acc[i][2 * q]     + bias[row0],     0.0f);
        float v1 = fmaxf(acc[i][2 * q + 1] + bias[row0 + 1], 0.0f);
        h1p[((size_t)b * 64 + (mtile * 8 + hi * 4 + q)) * 64 + col] = pk2(v0, v1);
      }
    }
  } else {
#pragma unroll
    for (int i = 0; i < 4; ++i) {
      int col = i * 16 + lm;
#pragma unroll
      for (int r = 0; r < 8; ++r) {
        int row = mtile * 16 + r + 8 * hi;
        scf[(size_t)b * 8192 + row * 64 + col] = acc[i][r] + bias[row];
      }
    }
  }
}

// =====================================================================
// K9: fused tail per batch: h = m1c2@h1 + b + shortcut -> LDS (packed bf16)
//     r1 = relu(m2c1@h + b) -> LDS; pred = m2c2@r1 + b -> (B,K,3) f32.
// =====================================================================
__global__ void k_final(const unsigned short* __restrict__ w1c2bf, const float* __restrict__ b1c2,
                       const float* __restrict__ scf, const unsigned* __restrict__ h1p,
                       const unsigned short* __restrict__ w2c1bf, const float* __restrict__ b2c1,
                       const float* __restrict__ w2c2, const float* __restrict__ b2c2,
                       float* __restrict__ out) {
  __shared__ unsigned hp[64 * 64];    // packed h (128 rows as 64 pair-rows x 64 cols)
  __shared__ unsigned r1p[32 * 64];   // packed r1 (64 rows as 32 pair-rows)
  int b = blockIdx.x;
  int w = threadIdx.x >> 5, lane = threadIdx.x & 31, hi = lane >> 4, lm = lane & 15;

  // ---- stage 1: h = w1c2 @ h1 + bias + shortcut (128x64) ----
  {
    int mtile = w;
    v8f acc[4] = {};
    const unsigned* arow = (const unsigned*)w1c2bf + (size_t)(mtile * 16 + lm) * 64;  // 128/2 dw
    const unsigned* hb = h1p + (size_t)b * 64 * 64;
    for (int kc = 0; kc < 4; ++kc) {
      FragAB a;
      a.q4[0] = *(const v4u*)(arow + kc * 16 + hi * 8);
      a.q4[1] = *(const v4u*)(arow + kc * 16 + hi * 8 + 4);
#pragma unroll
      for (int i = 0; i < 4; ++i) {
        FragAB bm;
#pragma unroll
        for (int j = 0; j < 8; ++j) bm.u[j] = hb[(size_t)(kc * 16 + j + 8 * hi) * 64 + i * 16 + lm];
        acc[i] = __builtin_amdgcn_wmma_f32_16x16x32_bf16(false, a.bf, false, bm.bf,
                                                         (short)0, acc[i], false, false);
      }
    }
#pragma unroll
    for (int i = 0; i < 4; ++i) {
      int col = i * 16 + lm;
#pragma unroll
      for (int q = 0; q < 4; ++q) {
        int row0 = mtile * 16 + 8 * hi + 2 * q;
        float v0 = acc[i][2 * q]     + b1c2[row0]     + scf[(size_t)b * 8192 + row0 * 64 + col];
        float v1 = acc[i][2 * q + 1] + b1c2[row0 + 1] + scf[(size_t)b * 8192 + (row0 + 1) * 64 + col];
        hp[(mtile * 8 + hi * 4 + q) * 64 + col] = pk2(v0, v1);
      }
    }
  }
  __syncthreads();

  // ---- stage 2: r1 = relu(w2c1 @ h + bias) (64x64), 16 tiles / 8 waves ----
  {
    int mtile = w >> 1;
    int ntb = (w & 1) * 2;
    v8f acc[2] = {};
    const unsigned* arow = (const unsigned*)w2c1bf + (size_t)(mtile * 16 + lm) * 64;
    for (int kc = 0; kc < 4; ++kc) {
      FragAB a;
      a.q4[0] = *(const v4u*)(arow + kc * 16 + hi * 8);
      a.q4[1] = *(const v4u*)(arow + kc * 16 + hi * 8 + 4);
#pragma unroll
      for (int i = 0; i < 2; ++i) {
        FragAB bm;
#pragma unroll
        for (int j = 0; j < 8; ++j) bm.u[j] = hp[(kc * 16 + j + 8 * hi) * 64 + (ntb + i) * 16 + lm];
        acc[i] = __builtin_amdgcn_wmma_f32_16x16x32_bf16(false, a.bf, false, bm.bf,
                                                         (short)0, acc[i], false, false);
      }
    }
#pragma unroll
    for (int i = 0; i < 2; ++i) {
      int col = (ntb + i) * 16 + lm;
#pragma unroll
      for (int q = 0; q < 4; ++q) {
        int row0 = mtile * 16 + 8 * hi + 2 * q;
        float v0 = fmaxf(acc[i][2 * q]     + b2c1[row0],     0.0f);
        float v1 = fmaxf(acc[i][2 * q + 1] + b2c1[row0 + 1], 0.0f);
        r1p[(mtile * 8 + hi * 4 + q) * 64 + col] = pk2(v0, v1);
      }
    }
  }
  __syncthreads();

  // ---- stage 3: pred (3x64) = w2c2 @ r1 + bias, write (B,K,3) ----
  if (threadIdx.x < 192) {
    int row = threadIdx.x >> 6, kk = threadIdx.x & 63;
    float acc = b2c2[row];
    for (int c = 0; c < 64; c += 2) {
      unsigned u = r1p[(c >> 1) * 64 + kk];
      acc = fmaf(w2c2[row * 64 + c],     bflo(u), acc);
      acc = fmaf(w2c2[row * 64 + c + 1], bfhi(u), acc);
    }
    out[(size_t)b * K * 3 + kk * 3 + row] = acc;
  }
}

// =====================================================================
// host launcher
// =====================================================================
extern "C" void kernel_launch(void* const* d_in, const int* in_sizes, int n_in,
                              void* d_out, int out_size, void* d_ws, size_t ws_size,
                              hipStream_t stream) {
  const float* pf     = (const float*)d_in[0];
  const float* gf     = (const float*)d_in[1];
  const float* ps1_w  = (const float*)d_in[2];
  const float* ps1_b  = (const float*)d_in[3];
  const float* m1c1_w = (const float*)d_in[4];
  const float* m1c1_b = (const float*)d_in[5];
  const float* m1c2_w = (const float*)d_in[6];
  const float* m1c2_b = (const float*)d_in[7];
  const float* m1sc_w = (const float*)d_in[8];
  const float* m1sc_b = (const float*)d_in[9];
  const float* m2c1_w = (const float*)d_in[10];
  const float* m2c1_b = (const float*)d_in[11];
  const float* m2c2_w = (const float*)d_in[12];
  const float* m2c2_b = (const float*)d_in[13];
  float* out = (float*)d_out;

  // ---- workspace layout (~164 MB; 256B-aligned chunks) ----
  char* ws = (char*)d_ws;
  size_t o = 0;
  unsigned short* xb   = (unsigned short*)(ws + o); o += (size_t)B * N * C * 2;      // 134.2 MB (L2-resident)
  unsigned short* gfb  = (unsigned short*)(ws + o); o += (size_t)B * C * 2;
  unsigned*       wpsf = (unsigned*)      (ws + o); o += (size_t)(C / 2) * 8192 * 4; // 16.8 MB (frag-blocked)
  unsigned short* w1c1 = (unsigned short*)(ws + o); o += (size_t)128 * CIN * 2;
  unsigned short* w1sc = (unsigned short*)(ws + o); o += (size_t)128 * CIN * 2;
  unsigned short* w1c2 = (unsigned short*)(ws + o); o += (size_t)128 * 128 * 2;
  unsigned short* w2c1 = (unsigned short*)(ws + o); o += (size_t)64 * 128 * 2;
  float*          dists= (float*)         (ws + o); o += (size_t)B * N * 4;
  float*          partD= (float*)         (ws + o); o += (size_t)B * 16 * 4;
  int*            partI= (int*)           (ws + o); o += (size_t)B * 16 * 4;
  int*            cur  = (int*)           (ws + o); o += 256;
  int*            idxb = (int*)           (ws + o); o += (size_t)B * K * 4;
  float*          x1f  = (float*)         (ws + o); o += (size_t)B * 128 * 64 * 4;
  unsigned*       catp = (unsigned*)      (ws + o); o += (size_t)B * (CIN / 2) * K * 4; // 8.9 MB (frag-blocked)
  unsigned*       h1p  = (unsigned*)      (ws + o); o += (size_t)B * 64 * 64 * 4;
  float*          scf  = (float*)         (ws + o); o += (size_t)B * 128 * 64 * 4;
  (void)in_sizes; (void)n_in; (void)out_size; (void)ws_size;

  // ---- phase 0: conversions / packing ----
  k_transpose<<<dim3(N / 32, C / 32, B), dim3(32, 8), 0, stream>>>(pf, xb);
  k_cvt_bf16<<<(B * C + 255) / 256, 256, 0, stream>>>(gf, gfb, B * C);
  k_cvt_bf16<<<(128 * CIN + 255) / 256, 256, 0, stream>>>(m1c1_w, w1c1, 128 * CIN);
  k_cvt_bf16<<<(128 * CIN + 255) / 256, 256, 0, stream>>>(m1sc_w, w1sc, 128 * CIN);
  k_cvt_bf16<<<(128 * 128 + 255) / 256, 256, 0, stream>>>(m1c2_w, w1c2, 128 * 128);
  k_cvt_bf16<<<(64 * 128 + 255) / 256, 256, 0, stream>>>(m2c1_w, w2c1, 64 * 128);
  k_pack_wps<<<((C / 2) * 8192 + 255) / 256, 256, 0, stream>>>(ps1_w, wpsf);
  k_init<<<(B * N + 255) / 256, 256, 0, stream>>>(dists, cur, idxb);

  // ---- phase 1: 63 sequential FPS steps (bf16 point set lives in L2) ----
  const unsigned* xb32 = (const unsigned*)xb;
  for (int t = 1; t < K; ++t) {
    k_fps_dist<<<dim3(16, B), 256, 0, stream>>>(xb32, cur, dists, partD, partI);
    k_fps_argmax<<<B, 32, 0, stream>>>(partD, partI, cur, idxb, t);
  }

  // ---- phase 2: WMMA GEMM pipeline ----
  k_gemm1<<<32, 256, 0, stream>>>(gfb, wpsf, ps1_b, x1f);
  k_pack_cat<<<(B * (CIN / 2) * K + 255) / 256, 256, 0, stream>>>(x1f, xb, gfb, idxb, catp);
  k_gemm2<<<B, 256, 0, stream>>>(w1c1, m1c1_b, catp, h1p, scf, 0);
  k_gemm2<<<B, 256, 0, stream>>>(w1sc, m1sc_b, catp, h1p, scf, 1);
  k_final<<<B, 256, 0, stream>>>(w1c2, m1c2_b, scf, h1p, w2c1, m2c1_b, m2c2_w, m2c2_b, out);
}
